// CausalShapedAttention_11278584120019
// MI455X (gfx1250) — matrix operand order repaired and checked by reading, not compiled, not run on hardware
//
#include <hip/hip_runtime.h>
#include <hip/hip_bf16.h>

// ---------------------------------------------------------------------------
// Shaped causal attention for MI455X (gfx1250, wave32, WMMA).
//   y = beta*softmax(QK^T/sqrt(D), causal)@V + alpha*V - gamma*(MC@V)
// MC@V decomposed into suffix means (O(T*D)).  All matmuls via
// v_wmma_f32_16x16x32_f16 (f16 operands, f32 accumulate).
// ---------------------------------------------------------------------------

typedef __attribute__((ext_vector_type(16))) _Float16 v16h;
typedef __attribute__((ext_vector_type(8)))  _Float16 v8h;
typedef __attribute__((ext_vector_type(4)))  _Float16 v4h;
typedef __attribute__((ext_vector_type(8)))  float    v8f;
typedef __attribute__((ext_vector_type(4)))  float    v4f;

static constexpr int B_ = 2, T_ = 2048, C_ = 1024, H_ = 16, D_ = 64;
static constexpr int M_  = B_ * T_;   // 4096 rows in projection GEMM
static constexpr int N2C = 2 * C_;    // 2048 cols (q then k)

union U16h { v16h v; v8h h[2]; };
__device__ inline v16h join8(v8h lo, v8h hi) { U16h u; u.h[0] = lo; u.h[1] = hi; return u.v; }
__device__ inline v8f v8f_zero() {
  v8f z;
  #pragma unroll
  for (int i = 0; i < 8; ++i) z[i] = 0.f;
  return z;
}

// ---------------------------------------------------------------------------
// Kernel 1: qk = x @ w^T  (M=4096, N=2048, K=1024), WMMA f16->f32.
// Output: qh (scaled by 1/sqrt(D)) and kh as f16 in [B,H,T,D] layout.
// ---------------------------------------------------------------------------
#define GEMM_BM 128
#define GEMM_BN 64
#define GEMM_BK 32
#define ASTR 40   // LDS row stride (halves); 80B keeps 16B alignment for b128 reads
#define BSTR 40

__global__ __launch_bounds__(256)
void qk_proj_kernel(const float* __restrict__ x, const float* __restrict__ w,
                    _Float16* __restrict__ qh, _Float16* __restrict__ kh)
{
  __shared__ _Float16 As[GEMM_BM * ASTR];
  __shared__ _Float16 Bs[GEMM_BN * BSTR];

  const int tid  = threadIdx.x;
  const int lane = tid & 31, wave = tid >> 5;
  const int lo   = lane & 15, hi = lane >> 4;
  const int n0   = blockIdx.x * GEMM_BN;
  const int m0   = blockIdx.y * GEMM_BM;
  const int kb8  = hi * 8, kb16 = hi * 16;

  v8f acc[4];
  #pragma unroll
  for (int i = 0; i < 4; ++i) acc[i] = v8f_zero();

  for (int k0 = 0; k0 < C_; k0 += GEMM_BK) {
    // --- batch ALL staging loads first so they are in flight concurrently
    v4f fa[4], fb[2];
    #pragma unroll
    for (int i = 0; i < 4; ++i) {
      int idx = tid + i * 256;            // 0..1023 : A tile 128x32
      int r = idx >> 3, cv = idx & 7;
      fa[i] = *(const v4f*)(x + (size_t)(m0 + r) * C_ + k0 + cv * 4);
    }
    #pragma unroll
    for (int i = 0; i < 2; ++i) {
      int idx = tid + i * 256;            // 0..511 : B tile 64x32
      int r = idx >> 3, cv = idx & 7;
      fb[i] = *(const v4f*)(w + (size_t)(n0 + r) * C_ + k0 + cv * 4);
    }
    if (k0 + GEMM_BK < C_) {              // hide next tile latency (global_prefetch_b8)
      __builtin_prefetch(x + (size_t)(m0 + (tid >> 1)) * C_ + k0 + GEMM_BK + (tid & 1) * 16, 0, 0);
      __builtin_prefetch(w + (size_t)(n0 + (tid & 63)) * C_ + k0 + GEMM_BK, 0, 0);
    }
    // --- convert f32->f16 and stage into LDS
    #pragma unroll
    for (int i = 0; i < 4; ++i) {
      int idx = tid + i * 256;
      int r = idx >> 3, cv = idx & 7;
      v4h hh = { (_Float16)fa[i][0], (_Float16)fa[i][1], (_Float16)fa[i][2], (_Float16)fa[i][3] };
      *(v4h*)&As[r * ASTR + cv * 4] = hh;
    }
    #pragma unroll
    for (int i = 0; i < 2; ++i) {
      int idx = tid + i * 256;
      int r = idx >> 3, cv = idx & 7;
      v4h hh = { (_Float16)fb[i][0], (_Float16)fb[i][1], (_Float16)fb[i][2], (_Float16)fb[i][3] };
      *(v4h*)&Bs[r * BSTR + cv * 4] = hh;
    }
    __syncthreads();

    // --- fragments: load A once and ALL B fragments, then issue WMMAs
    const int rowA = wave * 16 + lo;
    v16h a = join8(*(const v8h*)&As[rowA * ASTR + kb8],
                   *(const v8h*)&As[rowA * ASTR + 16 + kb8]);
    v16h bfr[4];
    #pragma unroll
    for (int nt = 0; nt < 4; ++nt) {
      const int rowB = nt * 16 + lo;      // column n of B = row n of w
      bfr[nt] = join8(*(const v8h*)&Bs[rowB * BSTR + kb8],
                      *(const v8h*)&Bs[rowB * BSTR + 16 + kb8]);
    }
    #pragma unroll
    for (int nt = 0; nt < 4; ++nt)
      acc[nt] = __builtin_amdgcn_wmma_f32_16x16x32_f16(
          false, a, false, bfr[nt], (short)0, acc[nt], false, false);
    __syncthreads();
  }

  // --- epilogue: scatter into [B,H,T,D] f16; fold 1/sqrt(D) into q
  const float qscale = 0.125f;            // 1/sqrt(64)
  #pragma unroll
  for (int nt = 0; nt < 4; ++nt) {
    const int n = n0 + nt * 16 + lo;
    #pragma unroll
    for (int j = 0; j < 8; ++j) {
      const int m = m0 + wave * 16 + j + 8 * hi;   // C-frag row mapping
      const int b = m / T_, t = m % T_;
      const float v = acc[nt][j];
      if (n < C_) {
        const int h = n >> 6, d = n & 63;
        qh[(((size_t)b * H_ + h) * T_ + t) * D_ + d] = (_Float16)(v * qscale);
      } else {
        const int nn = n - C_, h = nn >> 6, d = nn & 63;
        kh[(((size_t)b * H_ + h) * T_ + t) * D_ + d] = (_Float16)v;
      }
    }
  }
}

// ---------------------------------------------------------------------------
// Kernel 2: mc[b,h,i,d] = gamma * suffix-mean of V  (== gamma*(MC@V))
//   i <  T-1 : mean of x[b, i+1.., h*D+d]
//   i == T-1 : mean of all rows (softmax of an all-zero row is uniform 1/T)
// ---------------------------------------------------------------------------
__global__ void mc_kernel(const float* __restrict__ x,
                          const float* __restrict__ gamma_p,
                          float* __restrict__ mc)
{
  const int bh = blockIdx.x;              // 0..B*H-1
  const int b = bh / H_, h = bh % H_;
  const int d = threadIdx.x;              // 0..63
  const float gamma = *gamma_p;
  const float* xc = x + (size_t)b * T_ * C_ + h * D_ + d;
  float* mcc = mc + (size_t)bh * T_ * D_ + d;

  float total = 0.f;
  for (int j = 0; j < T_; ++j) total += xc[(size_t)j * C_];
  mcc[(size_t)(T_ - 1) * D_] = gamma * total * (1.0f / (float)T_);

  float S = 0.f;
  for (int i = T_ - 2; i >= 0; --i) {
    S += xc[(size_t)(i + 1) * C_];
    mcc[(size_t)i * D_] = gamma * S / (float)(T_ - 1 - i);
  }
}

// ---------------------------------------------------------------------------
// Kernel 3: causal flash attention, fused shaped-bias epilogue.
// Block = 256 threads (8 waves x 16 query rows = 128-row tile); key chunks of 32.
// ---------------------------------------------------------------------------
#define KSTR 72   // Kc row stride (halves): 144B, 16B-aligned v8h reads
#define VSTR 40   // Vt row stride (transposed V)
#define PSTR 40   // P staging row stride

__global__ __launch_bounds__(256)
void attn_kernel(const float* __restrict__ x, const _Float16* __restrict__ qh,
                 const _Float16* __restrict__ kh, const float* __restrict__ mc,
                 const float* __restrict__ alpha_p, const float* __restrict__ beta_p,
                 float* __restrict__ y)
{
  __shared__ _Float16 Kc[32 * KSTR];        // K chunk, row-major [key][d]
  __shared__ _Float16 Vt[64 * VSTR];        // V chunk, transposed [d][key]
  __shared__ _Float16 Pst[8 * 16 * PSTR];   // per-wave P staging (C-layout -> A-layout)

  const int tid  = threadIdx.x;
  const int lane = tid & 31, wave = tid >> 5;
  const int lo   = lane & 15, hi = lane >> 4;
  const int kb8  = hi * 8, kb16 = hi * 16;
  const int bh   = blockIdx.y, b = bh / H_, h = bh % H_;
  const int q0b  = blockIdx.x * 128;        // block's first query row
  const int q0   = q0b + wave * 16;         // this wave's first query row

  // Preload Q fragments (q already scaled by 1/sqrt(D)); A-layout: row=lo.
  const _Float16* qrow = qh + ((size_t)bh * T_ + q0 + lo) * D_;
  const v16h aQ0 = join8(*(const v8h*)(qrow + kb8),      *(const v8h*)(qrow + 16 + kb8));
  const v16h aQ1 = join8(*(const v8h*)(qrow + 32 + kb8), *(const v8h*)(qrow + 48 + kb8));

  float mrow[8], lrow[8];
  #pragma unroll
  for (int j = 0; j < 8; ++j) { mrow[j] = -1e30f; lrow[j] = 0.f; }
  v8f o[4];
  #pragma unroll
  for (int i = 0; i < 4; ++i) o[i] = v8f_zero();

  const int kend = q0b + 128;               // causal: keys < block's last row + 1
  for (int kb = 0; kb < kend; kb += 32) {
    // --- cooperative staging: batch all global loads, then convert/store
    const int kkK = tid >> 3, ddK = (tid & 7) * 8;
    const v8h kv = *(const v8h*)(kh + ((size_t)bh * T_ + kb + kkK) * D_ + ddK);
    const int kkV = tid & 31, d0 = (tid >> 5) * 8;
    const float* xp = x + ((size_t)b * T_ + kb + kkV) * C_ + h * D_ + d0;
    const v4f f0 = *(const v4f*)xp, f1 = *(const v4f*)(xp + 4);

    *(v8h*)&Kc[kkK * KSTR + ddK] = kv;
    #pragma unroll
    for (int i = 0; i < 4; ++i) Vt[(d0 + i) * VSTR + kkV]     = (_Float16)f0[i];
    #pragma unroll
    for (int i = 0; i < 4; ++i) Vt[(d0 + 4 + i) * VSTR + kkV] = (_Float16)f1[i];
    __syncthreads();

    if (kb <= q0 + 15) {                    // chunk fully masked for this wave? skip
      // --- load ALL K B-fragments, then issue the 4 score WMMAs back-to-back
      v16h bk[4];
      #pragma unroll
      for (int nt = 0; nt < 2; ++nt) {
        const int rowB = nt * 16 + lo;      // key (column of B) = row of Kc
        bk[nt * 2 + 0] = join8(*(const v8h*)&Kc[rowB * KSTR + kb8],
                               *(const v8h*)&Kc[rowB * KSTR + 16 + kb8]);
        bk[nt * 2 + 1] = join8(*(const v8h*)&Kc[rowB * KSTR + 32 + kb8],
                               *(const v8h*)&Kc[rowB * KSTR + 48 + kb8]);
      }
      v8f c0 = v8f_zero(), c1 = v8f_zero();
      c0 = __builtin_amdgcn_wmma_f32_16x16x32_f16(false, aQ0, false, bk[0], (short)0, c0, false, false);
      c1 = __builtin_amdgcn_wmma_f32_16x16x32_f16(false, aQ0, false, bk[2], (short)0, c1, false, false);
      c0 = __builtin_amdgcn_wmma_f32_16x16x32_f16(false, aQ1, false, bk[1], (short)0, c0, false, false);
      c1 = __builtin_amdgcn_wmma_f32_16x16x32_f16(false, aQ1, false, bk[3], (short)0, c1, false, false);

      float sc[2][8], p[2][8], cmax[8];
      #pragma unroll
      for (int j = 0; j < 8; ++j) {
        const int row = q0 + j + 8 * hi;
        const int key0 = kb + lo, key1 = kb + 16 + lo;
        const float s0 = (key0 <= row) ? c0[j] : -3.0e38f;  // causal mask
        const float s1 = (key1 <= row) ? c1[j] : -3.0e38f;
        sc[0][j] = s0; sc[1][j] = s1;
        cmax[j] = fmaxf(s0, s1);
      }

      // --- online softmax: row stats reduced across the 16-lane group
      #pragma unroll
      for (int j = 0; j < 8; ++j) {
        float m = cmax[j];
        m = fmaxf(m, __shfl_xor(m, 1, 32));
        m = fmaxf(m, __shfl_xor(m, 2, 32));
        m = fmaxf(m, __shfl_xor(m, 4, 32));
        m = fmaxf(m, __shfl_xor(m, 8, 32));
        const float mnew  = fmaxf(mrow[j], m);
        const float scale = __expf(mrow[j] - mnew);
        const float p0 = __expf(sc[0][j] - mnew);
        const float p1 = __expf(sc[1][j] - mnew);
        p[0][j] = p0; p[1][j] = p1;
        float ps = p0 + p1;
        ps += __shfl_xor(ps, 1, 32);
        ps += __shfl_xor(ps, 2, 32);
        ps += __shfl_xor(ps, 4, 32);
        ps += __shfl_xor(ps, 8, 32);
        lrow[j] = lrow[j] * scale + ps;
        mrow[j] = mnew;
        #pragma unroll
        for (int dt = 0; dt < 4; ++dt) o[dt][j] *= scale;
      }

      // --- restage P (C-layout) through per-wave LDS to A-layout; same-wave
      //     DS ops are in-order, so no barrier is required.
      _Float16* ps = &Pst[wave * 16 * PSTR];
      #pragma unroll
      for (int nt = 0; nt < 2; ++nt) {
        #pragma unroll
        for (int j = 0; j < 8; ++j)
          ps[(j + 8 * hi) * PSTR + nt * 16 + lo] = (_Float16)p[nt][j];
      }
      v16h aP = join8(*(const v8h*)&ps[lo * PSTR + kb8],
                      *(const v8h*)&ps[lo * PSTR + 16 + kb8]);

      // --- O += P @ V: load ALL V B-fragments, then 4 WMMAs back-to-back
      v16h bv[4];
      #pragma unroll
      for (int dt = 0; dt < 4; ++dt) {
        const int dcol = dt * 16 + lo;
        bv[dt] = join8(*(const v8h*)&Vt[dcol * VSTR + kb8],
                       *(const v8h*)&Vt[dcol * VSTR + 16 + kb8]);
      }
      #pragma unroll
      for (int dt = 0; dt < 4; ++dt)
        o[dt] = __builtin_amdgcn_wmma_f32_16x16x32_f16(
            false, aP, false, bv[dt], (short)0, o[dt], false, false);
    }
    __syncthreads();
  }

  // --- epilogue: y = beta*O/l + alpha*x - mc  (all f32)
  const float alpha = *alpha_p, beta = *beta_p;
  #pragma unroll
  for (int dt = 0; dt < 4; ++dt) {
    const int d = dt * 16 + lo;
    #pragma unroll
    for (int j = 0; j < 8; ++j) {
      const int t = q0 + j + 8 * hi;
      const float val = beta * o[dt][j] / lrow[j]
                      + alpha * x[((size_t)b * T_ + t) * C_ + h * D_ + d]
                      - mc[((size_t)bh * T_ + t) * D_ + d];
      y[((size_t)b * T_ + t) * C_ + h * D_ + d] = val;
    }
  }
}

// ---------------------------------------------------------------------------
extern "C" void kernel_launch(void* const* d_in, const int* in_sizes, int n_in,
                              void* d_out, int out_size, void* d_ws, size_t ws_size,
                              hipStream_t stream)
{
  const float* x       = (const float*)d_in[0];
  const float* w_attn  = (const float*)d_in[1];
  const float* alpha_p = (const float*)d_in[2];
  const float* beta_p  = (const float*)d_in[3];
  const float* gamma_p = (const float*)d_in[4];
  // d_in[5] = n_head (fixed 16, compiled in)
  float* y = (float*)d_out;

  char* ws = (char*)d_ws;
  const size_t nBHTD = (size_t)B_ * H_ * T_ * D_;        // 4 Mi elements
  _Float16* qh = (_Float16*)ws;                          //  8 MB
  _Float16* kh = (_Float16*)(ws + nBHTD * 2);            //  8 MB
  float*    mc = (float*)   (ws + nBHTD * 4);            // 16 MB  (total 32 MB)

  qk_proj_kernel<<<dim3(N2C / GEMM_BN, M_ / GEMM_BM), 256, 0, stream>>>(x, w_attn, qh, kh);
  mc_kernel<<<B_ * H_, D_, 0, stream>>>(x, gamma_p, mc);
  attn_kernel<<<dim3(T_ / 128, B_ * H_), 256, 0, stream>>>(x, qh, kh, mc, alpha_p, beta_p, y);
}
